// PromptSSM_69174743269727
// MI455X (gfx1250) — matrix operand order repaired and checked by reading, not compiled, not run on hardware
//
#include <hip/hip_runtime.h>
#include <hip/hip_bf16.h>
#include <math.h>

typedef _Float16 h8  __attribute__((ext_vector_type(8)));
typedef _Float16 h16 __attribute__((ext_vector_type(16)));
typedef float    f8  __attribute__((ext_vector_type(8)));
typedef int      i4  __attribute__((ext_vector_type(4)));
typedef __attribute__((address_space(1))) i4 gi4;  // global int4
typedef __attribute__((address_space(3))) i4 li4;  // LDS int4

union H16u { h16 v; h8 h[2]; };

#define BQ   16
#define CC   128
#define HH   64
#define WW   64
#define HWP  (HH*WW)     // 4096
#define KIM  (9*CC)      // 1152 im2col K

// LDS row strides (in halves), padded to dodge bank conflicts
#define ASTR 40
#define BSTR 136

// ---- async global->LDS copy (16B per lane), compile-safe fallback ----------
static __device__ inline void cp_async16(_Float16* dst, const _Float16* src) {
#if __has_builtin(__builtin_amdgcn_global_load_async_to_lds_b128)
  __builtin_amdgcn_global_load_async_to_lds_b128((gi4*)src, (li4*)dst, 0, 0);
#else
  *(h8*)dst = *(const h8*)src;
#endif
}
static __device__ inline void async_wait() {
#if __has_builtin(__builtin_amdgcn_global_load_async_to_lds_b128)
#if __has_builtin(__builtin_amdgcn_s_wait_asynccnt)
  __builtin_amdgcn_s_wait_asynccnt(0);
#else
  asm volatile("s_wait_asynccnt 0" ::: "memory");
#endif
#endif
}

// ---- WMMA fragment loaders (wave32 layouts per CDNA5 ISA 7.12.2) ----
// A 16x32 f16: lane L -> row M=L%16; halves: K in [8*hi,8*hi+8) then [16+8*hi,+8)
static __device__ inline h16 ldsA(const _Float16* base, int stride, int lane) {
  H16u u;
  const int m = lane & 15, hi = lane >> 4;
  const _Float16* p = base + m * stride;
  u.h[0] = *(const h8*)(p + 8 * hi);
  u.h[1] = *(const h8*)(p + 16 + 8 * hi);
  return u.v;
}
// B 32x16 f16: lane L -> col N=n0+L%16; halves: K in [8*hi,8*hi+8) then [16+8*hi,+8)
static __device__ inline h16 ldsB(const _Float16* base, int stride, int lane, int n0) {
  H16u u;
  const int hi = lane >> 4;
  const _Float16* p = base + n0 + (lane & 15);
#pragma unroll
  for (int s = 0; s < 8; ++s) {
    u.v[s] = p[(8 * hi + s) * stride];
    u.v[s + 8] = p[(16 + 8 * hi + s) * stride];
  }
  return u.v;
}
static __device__ inline f8 wmma_f16(h16 a, h16 b, f8 c) {
  return __builtin_amdgcn_wmma_f32_16x16x32_f16(false, a, false, b, (short)0, c,
                                                false, false);
}

// ---- weight repack: (N,Ci,3,3) f32 -> (K=1152, N=128) f16, k=(dy*3+dx)*128+ci
__global__ __launch_bounds__(256) void pack_weight(const float* __restrict__ w,
                                                   _Float16* __restrict__ Wp) {
  int idx = blockIdx.x * 256 + threadIdx.x;  // k*128 + n
  if (idx >= KIM * CC) return;
  int k = idx >> 7, n = idx & 127;
  int tap = k >> 7, ci = k & 127;
  int dy = tap / 3, dx = tap % 3;
  Wp[idx] = (_Float16)w[(((size_t)(n * CC + ci)) * 3 + dy) * 3 + dx];
}

// ---- NCHW f32 -> NHWC f16, tiled LDS transpose (coalesced both sides) -------
__global__ __launch_bounds__(256) void nchw_to_nhwc_h(const float* __restrict__ x,
                                                      _Float16* __restrict__ xh) {
  __shared__ float t[32][33];
  const int bid = blockIdx.x;
  const int ct = bid & 3, pt = (bid >> 2) & 127, b = bid >> 9;
  const int c0 = ct * 32, p0 = pt * 32;
  const int j = threadIdx.x & 31, i0 = threadIdx.x >> 5;
#pragma unroll
  for (int s = 0; s < 4; ++s) {
    const int c = i0 + s * 8;
    t[c][j] = x[((size_t)(b * CC + c0 + c)) * HWP + p0 + j];
  }
  __syncthreads();
#pragma unroll
  for (int s = 0; s < 4; ++s) {
    const int p = i0 + s * 8;
    xh[((size_t)(b * HWP + p0 + p)) * CC + c0 + j] = (_Float16)t[j][p];
  }
}

// ---- NHWC f32 -> NCHW f32 (final output layout), tiled transpose ------------
__global__ __launch_bounds__(256) void nhwc_to_nchw_f(const float* __restrict__ src,
                                                      float* __restrict__ dst) {
  __shared__ float t[32][33];
  const int bid = blockIdx.x;
  const int ct = bid & 3, pt = (bid >> 2) & 127, b = bid >> 9;
  const int c0 = ct * 32, p0 = pt * 32;
  const int j = threadIdx.x & 31, i0 = threadIdx.x >> 5;
#pragma unroll
  for (int s = 0; s < 4; ++s) {
    const int p = i0 + s * 8;
    t[p][j] = src[((size_t)(b * HWP + p0 + p)) * CC + c0 + j];
  }
  __syncthreads();
#pragma unroll
  for (int s = 0; s < 4; ++s) {
    const int c = i0 + s * 8;
    dst[((size_t)(b * CC + c0 + c)) * HWP + p0 + j] = t[j][c];
  }
}

// ---- implicit-im2col 3x3 SAME conv as WMMA GEMM -----------------------------
// Block tile: M=128 spatial (2 rows y0,y0+1) x N=128 chan; wave tile 32x64.
// MODE 0: out_h = conv+bias (f16 NHWC)        [Q/K/V]
// MODE 1: out_h = gelu(conv+bias) (f16 NHWC)  [f1]
// MODE 2: outf  = resid + conv+bias, f32 NHWC in-place [f2 + residual]
template <int MODE>
__global__ __launch_bounds__(256)
void conv3x3_wmma(const _Float16* __restrict__ xh, const _Float16* __restrict__ Wp,
                  const float* __restrict__ bias, _Float16* __restrict__ outh,
                  const float* __restrict__ resid, float* __restrict__ outf) {
  __shared__ _Float16 Al[128 * ASTR];
  __shared__ _Float16 Bl[32 * BSTR];
  const int tid = threadIdx.x, lane = tid & 31, wave = tid >> 5;
  const int b  = blockIdx.x / (HH / 2);
  const int y0 = (blockIdx.x % (HH / 2)) * 2;
  const int m0  = (wave >> 1) * 32;  // two 16-row A sub-tiles
  const int n0w = (wave & 1) * 64;   // channel half

  f8 acc[2][4] = {};
  const int arow = tid >> 2;        // 0..63 (this thread also does arow+64)
  const int acol = (tid & 3) * 8;   // k offset within 32-chunk
  const int brow = tid >> 3;        // 0..31
  const int bcol = (tid & 7) * 16;

  for (int kk = 0; kk < KIM; kk += 32) {
    const int tap = kk >> 7;                    // one tap per 32-chunk (32 | 128)
    const int dy = tap / 3 - 1, dx = tap % 3 - 1;
    // B stage: packed weights via async DMA to LDS
    {
      const _Float16* src = Wp + (size_t)(kk + brow) * CC + bcol;
      cp_async16(Bl + brow * BSTR + bcol, src);
      cp_async16(Bl + brow * BSTR + bcol + 8, src + 8);
    }
    // A stage: bounds-checked im2col gather, two spatial rows per thread
#pragma unroll
    for (int half = 0; half < 2; ++half) {
      const int rr = arow + 64 * half;          // 0..127
      const int ly = rr >> 6, x = rr & 63;
      const int yy = y0 + ly + dy, xx = x + dx;
      h8 av = {};
      if ((unsigned)yy < HH && (unsigned)xx < WW)
        av = *(const h8*)(xh + ((size_t)(b * HWP + yy * WW + xx)) * CC +
                          (kk & 127) + acol);
      *(h8*)(Al + rr * ASTR + acol) = av;
    }
    async_wait();
    __syncthreads();
    h16 a0 = ldsA(Al + m0 * ASTR, ASTR, lane);
    h16 a1 = ldsA(Al + (m0 + 16) * ASTR, ASTR, lane);
#pragma unroll
    for (int t = 0; t < 4; ++t) {
      h16 bf = ldsB(Bl, BSTR, lane, n0w + 16 * t);
      acc[0][t] = wmma_f16(a0, bf, acc[0][t]);
      acc[1][t] = wmma_f16(a1, bf, acc[1][t]);
    }
    __syncthreads();
  }
  const int hi = lane >> 4, nl = lane & 15;
#pragma unroll
  for (int mi = 0; mi < 2; ++mi)
#pragma unroll
    for (int t = 0; t < 4; ++t) {
      const int n = n0w + 16 * t + nl;
      const float bn = bias[n];
#pragma unroll
      for (int r = 0; r < 8; ++r) {
        const int rr = m0 + mi * 16 + (hi << 3) + r;  // C/D: M = r + 8*hi
        const int ly = rr >> 6, x = rr & 63;
        float v = acc[mi][t][r] + bn;
        const size_t pidx = (size_t)(b * HWP + (y0 + ly) * WW + x);
        if (MODE == 0) {
          outh[pidx * CC + n] = (_Float16)v;
        } else if (MODE == 1) {
          outh[pidx * CC + n] =
              (_Float16)(0.5f * v * (1.0f + erff(v * 0.70710678f)));
        } else {
          outf[pidx * CC + n] = resid[pidx * CC + n] + v;  // in-place residual
        }
      }
    }
}

// ---- attn logits: logits[b,c,d] = (1/64) * sum_p K[b,p,c] * Q[b,p,d] --------
__global__ __launch_bounds__(256)
void attn_logits_wmma(const _Float16* __restrict__ Kh, const _Float16* __restrict__ Qh,
                      float* __restrict__ logits) {
  __shared__ _Float16 Al[128 * ASTR];  // K transposed: [c][p]
  __shared__ _Float16 Bl[32 * BSTR];   // Q: [p][d]
  const int tid = threadIdx.x, lane = tid & 31, wave = tid >> 5;
  const int b = blockIdx.x;
  const int m0 = wave * 16;
  f8 acc[8] = {};
  const int prow = tid >> 3;       // 0..31
  const int c0 = (tid & 7) * 16;

  for (int pp = 0; pp < HWP; pp += 32) {
    const size_t srow = ((size_t)(b * HWP + pp + prow)) * CC + c0;
    H16u kv;
    kv.h[0] = *(const h8*)(Kh + srow);
    kv.h[1] = *(const h8*)(Kh + srow + 8);
#pragma unroll
    for (int j = 0; j < 16; ++j) Al[(c0 + j) * ASTR + prow] = kv.v[j];
    cp_async16(Bl + prow * BSTR + c0, Qh + srow);
    cp_async16(Bl + prow * BSTR + c0 + 8, Qh + srow + 8);
    async_wait();
    __syncthreads();
    h16 a = ldsA(Al + m0 * ASTR, ASTR, lane);
#pragma unroll
    for (int t = 0; t < 8; ++t)
      acc[t] = wmma_f16(a, ldsB(Bl, BSTR, lane, 16 * t), acc[t]);
    __syncthreads();
  }
  const int hi = lane >> 4, nl = lane & 15;
#pragma unroll
  for (int t = 0; t < 8; ++t)
#pragma unroll
    for (int r = 0; r < 8; ++r) {
      const int c = m0 + (hi << 3) + r, d = 16 * t + nl;
      logits[((size_t)(b * CC + c)) * CC + d] = acc[t][r] * (1.0f / 64.0f);
    }
}

// ---- in-place row softmax over 128 cols -------------------------------------
__global__ __launch_bounds__(128) void softmax128(float* __restrict__ a) {
  __shared__ float red[128];
  const int row = blockIdx.x, t = threadIdx.x;
  float v = a[(size_t)row * CC + t];
  red[t] = v; __syncthreads();
  for (int s = 64; s > 0; s >>= 1) { if (t < s) red[t] = fmaxf(red[t], red[t + s]); __syncthreads(); }
  const float mx = red[0]; __syncthreads();
  const float e = expf(v - mx);
  red[t] = e; __syncthreads();
  for (int s = 64; s > 0; s >>= 1) { if (t < s) red[t] += red[t + s]; __syncthreads(); }
  a[(size_t)row * CC + t] = e / red[0];
}

// ---- M[b,c,d] = sum_e Pspe[b,c,e] * A[b,e,d] (tiny), store f16 --------------
__global__ __launch_bounds__(256)
void prompt_fuse(const float* __restrict__ Pspe, const float* __restrict__ A,
                 _Float16* __restrict__ Mh) {
  const int idx = blockIdx.x * 256 + threadIdx.x;  // b*C*C
  const int b = idx >> 14, c = (idx >> 7) & 127, d = idx & 127;
  const float* pr = Pspe + ((size_t)(b * CC + c)) * CC;
  const float* ac = A + (size_t)b * CC * CC + d;
  float s = 0.f;
#pragma unroll 4
  for (int e = 0; e < CC; ++e) s += pr[e] * ac[(size_t)e * CC];
  Mh[idx] = (_Float16)s;
}

// ---- F_ssm[b,p,d] = sum_c V[b,p,c] * M[b,c,d]; store f32 + f16 NHWC ---------
__global__ __launch_bounds__(256)
void f3_wmma(const _Float16* __restrict__ Vh, const _Float16* __restrict__ Mh,
             float* __restrict__ Ff, _Float16* __restrict__ Fh) {
  __shared__ _Float16 Al[64 * ASTR];
  __shared__ _Float16 Bl[32 * BSTR];
  const int tid = threadIdx.x, lane = tid & 31, wave = tid >> 5;
  const int b = blockIdx.x >> 6;
  const int p0 = (blockIdx.x & 63) * 64;
  const int m0 = (wave >> 1) * 16, n0w = (wave & 1) * 64;
  f8 acc[4] = {};
  const int arow = tid >> 2, acol = (tid & 3) * 8;
  const int brow = tid >> 3, bcol = (tid & 7) * 16;

  for (int kk = 0; kk < CC; kk += 32) {
    cp_async16(Al + arow * ASTR + acol,
               Vh + ((size_t)(b * HWP + p0 + arow)) * CC + kk + acol);
    const _Float16* src = Mh + ((size_t)(b * CC + kk + brow)) * CC + bcol;
    cp_async16(Bl + brow * BSTR + bcol, src);
    cp_async16(Bl + brow * BSTR + bcol + 8, src + 8);
    async_wait();
    __syncthreads();
    h16 a = ldsA(Al + m0 * ASTR, ASTR, lane);
#pragma unroll
    for (int t = 0; t < 4; ++t)
      acc[t] = wmma_f16(a, ldsB(Bl, BSTR, lane, n0w + 16 * t), acc[t]);
    __syncthreads();
  }
  const int hi = lane >> 4, nl = lane & 15;
#pragma unroll
  for (int t = 0; t < 4; ++t)
#pragma unroll
    for (int r = 0; r < 8; ++r) {
      const int p = p0 + m0 + (hi << 3) + r, n = n0w + 16 * t + nl;
      const float v = acc[t][r];
      const size_t o = ((size_t)(b * HWP) + p) * CC + n;
      Ff[o] = v;
      Fh[o] = (_Float16)v;
    }
}

extern "C" void kernel_launch(void* const* d_in, const int* in_sizes, int n_in,
                              void* d_out, int out_size, void* d_ws, size_t ws_size,
                              hipStream_t stream) {
  (void)in_sizes; (void)n_in; (void)out_size; (void)ws_size;
  const float* F_in = (const float*)d_in[0];
  const float* Pspe = (const float*)d_in[1];
  const float* w5[5] = {(const float*)d_in[2], (const float*)d_in[4],
                        (const float*)d_in[6], (const float*)d_in[8],
                        (const float*)d_in[10]};
  const float* b5[5] = {(const float*)d_in[3], (const float*)d_in[5],
                        (const float*)d_in[7], (const float*)d_in[9],
                        (const float*)d_in[11]};
  float* out = (float*)d_out;
  char* ws = (char*)d_ws;

  size_t off = 0;
  auto alloc = [&](size_t bytes) { size_t o = off; off += (bytes + 255) & ~(size_t)255; return o; };
  const size_t xh_o = alloc((size_t)BQ * HWP * CC * 2);
  size_t wp_o[5]; for (int i = 0; i < 5; ++i) wp_o[i] = alloc((size_t)KIM * CC * 2);
  const size_t qh_o = alloc((size_t)BQ * HWP * CC * 2);
  const size_t kh_o = alloc((size_t)BQ * HWP * CC * 2);
  const size_t vh_o = alloc((size_t)BQ * HWP * CC * 2);
  const size_t lg_o = alloc((size_t)BQ * CC * CC * 4);
  const size_t mh_o = alloc((size_t)BQ * CC * CC * 2);
  const size_t ff_o = alloc((size_t)BQ * HWP * CC * 4);

  _Float16* xh = (_Float16*)(ws + xh_o);
  _Float16* Wp[5]; for (int i = 0; i < 5; ++i) Wp[i] = (_Float16*)(ws + wp_o[i]);
  _Float16* Qh = (_Float16*)(ws + qh_o);
  _Float16* Kh = (_Float16*)(ws + kh_o);
  _Float16* Vh = (_Float16*)(ws + vh_o);
  float*    Lg = (float*)(ws + lg_o);
  _Float16* Mh = (_Float16*)(ws + mh_o);
  float*    Ff = (float*)(ws + ff_o);
  _Float16* Fh = Qh;  // reuse after attention
  _Float16* Gh = Kh;  // reuse after attention

  for (int i = 0; i < 5; ++i)
    pack_weight<<<(KIM * CC + 255) / 256, 256, 0, stream>>>(w5[i], Wp[i]);
  nchw_to_nhwc_h<<<BQ * 128 * 4, 256, 0, stream>>>(F_in, xh);

  conv3x3_wmma<0><<<BQ * (HH / 2), 256, 0, stream>>>(xh, Wp[0], b5[0], Qh, nullptr, nullptr);
  conv3x3_wmma<0><<<BQ * (HH / 2), 256, 0, stream>>>(xh, Wp[1], b5[1], Kh, nullptr, nullptr);
  conv3x3_wmma<0><<<BQ * (HH / 2), 256, 0, stream>>>(xh, Wp[2], b5[2], Vh, nullptr, nullptr);

  attn_logits_wmma<<<BQ, 256, 0, stream>>>(Kh, Qh, Lg);
  softmax128<<<BQ * CC, 128, 0, stream>>>(Lg);
  prompt_fuse<<<(BQ * CC * CC) / 256, 256, 0, stream>>>(Pspe, Lg, Mh);
  f3_wmma<<<BQ * (HWP / 64), 256, 0, stream>>>(Vh, Mh, Ff, Fh);

  conv3x3_wmma<1><<<BQ * (HH / 2), 256, 0, stream>>>(Fh, Wp[3], b5[3], Gh, nullptr, nullptr);
  conv3x3_wmma<2><<<BQ * (HH / 2), 256, 0, stream>>>(Gh, Wp[4], b5[4], nullptr, Ff, Ff);

  nhwc_to_nchw_f<<<BQ * 128 * 4, 256, 0, stream>>>(Ff, out);
}
